// Split_SNLB_46145128628250
// MI455X (gfx1250) — matrix operand order repaired and checked by reading, not compiled, not run on hardware
//
#include <hip/hip_runtime.h>
#include <hip/hip_bf16.h>

typedef __attribute__((ext_vector_type(16))) __bf16 v16bf;
typedef __attribute__((ext_vector_type(8)))  __bf16 v8bf;
typedef __attribute__((ext_vector_type(8)))  float  v8f;
typedef __attribute__((ext_vector_type(4)))  float  v4f;

#define LOG2E 1.4426950408889634f

// ---- LDS layout (bytes). Strides padded so A/B fragment loads hit banks 4L mod 64.
#define XS_S 264   // x chunk: [128 pix][264] bf16
#define WB_S 264   // weight:  [128 i ][264] bf16
#define G_S  264   // g:       [128 i ][264] bf16
#define PHI_S 136  // phi:     [256 k ][136] bf16
#define TC_S 136   // theta/y: [128 q ][136] bf16
#define PS_S 40    // per-wave p scratch: [16][40] bf16

#define OFF_PHI 0
#define OFF_G   69632
#define OFF_W   137216
#define OFF_XS  204800
#define OFF_TC  272384
#define OFF_PS  307200
#define OFF_PAR 317440
#define LDS_BYTES 321024   // <= 320KB WGP LDS

static __device__ __forceinline__ v16bf cat16(v8bf lo, v8bf hi) {
  return __builtin_shufflevector(lo, hi, 0,1,2,3,4,5,6,7,8,9,10,11,12,13,14,15);
}
// A fragment (16x32 bf16): lane<16 -> K = kb+{0..7, 16..23}; lane>=16 -> +8
static __device__ __forceinline__ v16bf ldsA(const __bf16* row, int kb, int lane) {
  int off = kb + ((lane & 16) ? 8 : 0);
  v8bf lo = *(const v8bf*)(row + off);
  v8bf hi = *(const v8bf*)(row + off + 16);
  return cat16(lo, hi);
}
// B fragment (32x16 bf16): lane<16 -> K = kb+{0..7, 16..23}; lane>=16 -> +8
static __device__ __forceinline__ v16bf ldsB(const __bf16* row, int kb, int lane) {
  int off = kb + ((lane & 16) ? 8 : 0);
  v8bf lo = *(const v8bf*)(row + off);
  v8bf hi = *(const v8bf*)(row + off + 16);
  return cat16(lo, hi);
}
#define WMMA(a,b,c) __builtin_amdgcn_wmma_f32_16x16x32_bf16(false,(a),false,(b),(short)0,(c),false,false)

#define V8F_ZERO {0.f,0.f,0.f,0.f,0.f,0.f,0.f,0.f}

__global__ __launch_bounds__(256, 1)
void snlb_kernel(const float* __restrict__ x,
                 const float* __restrict__ theta_w, const float* __restrict__ theta_b,
                 const float* __restrict__ phi_w,   const float* __restrict__ phi_b,
                 const float* __restrict__ g_w,     const float* __restrict__ g_b,
                 const float* __restrict__ w_w,     const float* __restrict__ w_b,
                 const float* __restrict__ bn_g,    const float* __restrict__ bn_b,
                 const float* __restrict__ bn_m,    const float* __restrict__ bn_v,
                 float* __restrict__ out) {
  extern __shared__ char smem[];
  __bf16* ph  = (__bf16*)(smem + OFF_PHI);
  __bf16* gl  = (__bf16*)(smem + OFF_G);
  __bf16* wb  = (__bf16*)(smem + OFF_W);
  __bf16* xs  = (__bf16*)(smem + OFF_XS);
  __bf16* tc  = (__bf16*)(smem + OFF_TC);   // theta chunk, later reused as y
  __bf16* psb = (__bf16*)(smem + OFF_PS);
  float*  tbL = (float*)(smem + OFF_PAR);
  float*  pbL = tbL + 128;
  float*  gbL = pbL + 128;
  float*  scL = gbL + 128;
  float*  shL = scL + 256;

  const int tid  = threadIdx.x;
  const int lane = tid & 31;
  const int wv   = tid >> 5;          // wave id 0..7
  const int hh   = (lane >> 4) & 1;   // lane half
  const int l15  = lane & 15;

  const int blk = blockIdx.x;         // 128 tiles
  const int b   = blk >> 4;
  const int bi  = (blk >> 2) & 3;
  const int bj  = blk & 3;

  const float* xtile = x + (size_t)b * 4194304 + bi * (32 * 128) + bj * 32; // + c*16384 + r*128 + col
  float* outp = out + (size_t)b * 4194304 + (size_t)((bi * 4 + bj) * 256) * 1024;

  // precompute small params into LDS
  if (tid < 128) { tbL[tid] = theta_b[tid]; pbL[tid] = phi_b[tid]; gbL[tid] = g_b[tid]; }
  { float s = bn_g[tid] * rsqrtf(bn_v[tid] + 1e-5f);
    scL[tid] = s; shL[tid] = bn_b[tid] + (w_b[tid] - bn_m[tid]) * s; }
  __syncthreads();

  // ================= Phase 1: pooled phi [k][i] and g [i][k] =================
  for (int t = 0; t < 8; ++t) {
    // stage x rows 4t..4t+3 as bf16 [pixel][channel]
    for (int seg = wv; seg < 1024; seg += 8) {
      int c = seg >> 2, rl = seg & 3;
      float v = xtile[c * 16384 + (4 * t + rl) * 128 + lane];
      xs[(rl * 32 + lane) * XS_S + c] = (__bf16)v;
    }
    __syncthreads();
    for (int wsel = 0; wsel < 2; ++wsel) {
      const float* wsrc = wsel ? g_w : phi_w;
      for (int seg = wv; seg < 1024; seg += 8) {
        int i = seg >> 3, cb = seg & 7;
        wb[i * WB_S + cb * 32 + lane] = (__bf16)wsrc[i * 256 + cb * 32 + lane];
      }
      __syncthreads();
      v16bf aF[8];
      { const __bf16* wrow = wb + (16 * wv + l15) * WB_S;
        #pragma unroll
        for (int kk = 0; kk < 8; ++kk) aF[kk] = ldsA(wrow, kk * 32, lane); }
      #pragma unroll
      for (int u = 0; u < 2; ++u)
      #pragma unroll
      for (int s2 = 0; s2 < 2; ++s2) {
        v8f Da = V8F_ZERO, Db = V8F_ZERO;
        const __bf16* xrA = xs + ((4 * u + s2) * 16 + l15) * XS_S;       // row r=2u
        const __bf16* xrB = xs + ((4 * u + 2 + s2) * 16 + l15) * XS_S;   // row r=2u+1
        #pragma unroll
        for (int kk = 0; kk < 8; ++kk) {
          Da = WMMA(aF[kk], ldsB(xrA, kk * 32, lane), Da);
          Db = WMMA(aF[kk], ldsB(xrB, kk * 32, lane), Db);
        }
        #pragma unroll
        for (int v = 0; v < 8; ++v) {   // 2x2 maxpool: rows in-lane, cols via shfl_xor(1)
          float m1 = fmaxf(Da[v], Db[v]);
          m1 = fmaxf(m1, __shfl_xor(m1, 1, 32));
          if (!(lane & 1)) {
            int k = (2 * t + u) * 16 + s2 * 8 + (l15 >> 1);
            int i = 16 * wv + v + 8 * hh;
            if (wsel) gl[i * G_S + k]   = (__bf16)(m1 + gbL[i]);
            else      ph[k * PHI_S + i] = (__bf16)(m1 + pbL[i]);
          }
        }
      }
      __syncthreads();
    }
  }

  // theta_w resident in wb for all of phase 2
  for (int seg = wv; seg < 1024; seg += 8) {
    int i = seg >> 3, cb = seg & 7;
    wb[i * WB_S + cb * 32 + lane] = (__bf16)theta_w[i * 256 + cb * 32 + lane];
  }
  __syncthreads();

  // ================= Phase 2: theta -> flash softmax*g -> out conv =================
  __bf16* ps = psb + wv * 16 * PS_S;
  for (int t = 0; t < 8; ++t) {
    for (int seg = wv; seg < 1024; seg += 8) {
      int c = seg >> 2, rl = seg & 3;
      float v = xtile[c * 16384 + (4 * t + rl) * 128 + lane];
      xs[(rl * 32 + lane) * XS_S + c] = (__bf16)v;
    }
    __syncthreads();

    // theta conv, transposed store tc[q][i] (+bias)
    { v16bf aF[8];
      const __bf16* wrow = wb + (16 * wv + l15) * WB_S;
      #pragma unroll
      for (int kk = 0; kk < 8; ++kk) aF[kk] = ldsA(wrow, kk * 32, lane);
      for (int nt = 0; nt < 8; ++nt) {
        v8f d = V8F_ZERO;
        const __bf16* xrow = xs + (nt * 16 + l15) * XS_S;
        #pragma unroll
        for (int kk = 0; kk < 8; ++kk) d = WMMA(aF[kk], ldsB(xrow, kk * 32, lane), d);
        #pragma unroll
        for (int v = 0; v < 8; ++v) {
          int i = 16 * wv + v + 8 * hh;
          tc[(nt * 16 + l15) * TC_S + i] = (__bf16)(d[v] + tbL[i]);
        }
      }
    }
    __syncthreads();

    // flash: wave wv owns q rows [16wv,16wv+16)
    {
      v16bf thA[4];
      const __bf16* trow = tc + (16 * wv + l15) * TC_S;
      #pragma unroll
      for (int ik = 0; ik < 4; ++ik) thA[ik] = ldsA(trow, ik * 32, lane);
      v8f mrun, srun, yac[8];
      #pragma unroll
      for (int v = 0; v < 8; ++v) { mrun[v] = -3.0e38f; srun[v] = 0.f; }
      #pragma unroll
      for (int it = 0; it < 8; ++it) { v8f z = V8F_ZERO; yac[it] = z; }

      for (int s2 = 0; s2 < 8; ++s2) {       // 32 k's per step
        v8f f0 = V8F_ZERO, f1 = V8F_ZERO;
        const __bf16* p0r = ph + ((2 * s2) * 16 + l15) * PHI_S;
        const __bf16* p1r = ph + ((2 * s2 + 1) * 16 + l15) * PHI_S;
        #pragma unroll
        for (int ik = 0; ik < 4; ++ik) {
          f0 = WMMA(thA[ik], ldsB(p0r, ik * 32, lane), f0);
          f1 = WMMA(thA[ik], ldsB(p1r, ik * 32, lane), f1);
        }
        v8f scl, p0, p1;
        #pragma unroll
        for (int v = 0; v < 8; ++v) {
          float m1 = fmaxf(f0[v], f1[v]);
          #pragma unroll
          for (int msk = 1; msk < 16; msk <<= 1) m1 = fmaxf(m1, __shfl_xor(m1, msk, 32));
          float mn = fmaxf(mrun[v], m1);
          scl[v] = exp2f((mrun[v] - mn) * LOG2E);
          float a  = exp2f((f0[v] - mn) * LOG2E);
          float c2 = exp2f((f1[v] - mn) * LOG2E);
          p0[v] = a; p1[v] = c2;
          float sadd = a + c2;
          #pragma unroll
          for (int msk = 1; msk < 16; msk <<= 1) sadd += __shfl_xor(sadd, msk, 32);
          srun[v] = srun[v] * scl[v] + sadd;
          mrun[v] = mn;
        }
        #pragma unroll
        for (int it = 0; it < 8; ++it) yac[it] = yac[it] * scl;
        // D-layout -> A-layout via per-wave LDS scratch
        #pragma unroll
        for (int v = 0; v < 8; ++v) {
          int row = v + 8 * hh;
          ps[row * PS_S + l15]      = (__bf16)p0[v];
          ps[row * PS_S + 16 + l15] = (__bf16)p1[v];
        }
        v16bf pA = ldsA(ps + l15 * PS_S, 0, lane);
        #pragma unroll
        for (int it = 0; it < 8; ++it) {
          const __bf16* gr = gl + (it * 16 + l15) * G_S;
          yac[it] = WMMA(pA, ldsB(gr, s2 * 32, lane), yac[it]);
        }
      }
      #pragma unroll
      for (int v = 0; v < 8; ++v) srun[v] = 1.0f / srun[v];
      #pragma unroll
      for (int it = 0; it < 8; ++it)
        #pragma unroll
        for (int v = 0; v < 8; ++v)
          tc[(16 * wv + v + 8 * hh) * TC_S + it * 16 + l15] = (__bf16)(yac[it][v] * srun[v]);
    }
    __syncthreads();

    // out conv: D[o][q] = w2 . y^T  (+BN +residual), coalesced over q
    #pragma unroll
    for (int oo = 0; oo < 2; ++oo) {
      int ot = wv + 8 * oo;
      v16bf wA[4];
      const float* wr = w_w + (ot * 16 + l15) * 128;   // w_w is [256][128], K=i contiguous
      #pragma unroll
      for (int ik = 0; ik < 4; ++ik) {
        int off = ik * 32 + ((lane & 16) ? 8 : 0);
        v4f a0 = *(const v4f*)(wr + off);
        v4f a1 = *(const v4f*)(wr + off + 4);
        v4f b0 = *(const v4f*)(wr + off + 16);
        v4f b1 = *(const v4f*)(wr + off + 20);
        v16bf r;
        #pragma unroll
        for (int e = 0; e < 4; ++e) {
          r[e] = (__bf16)a0[e]; r[4 + e] = (__bf16)a1[e];
          r[8 + e] = (__bf16)b0[e]; r[12 + e] = (__bf16)b1[e];
        }
        wA[ik] = r;
      }
      for (int nt = 0; nt < 8; ++nt) {
        v8f d = V8F_ZERO;
        const __bf16* yr = tc + (nt * 16 + l15) * TC_S;
        #pragma unroll
        for (int ik = 0; ik < 4; ++ik) d = WMMA(wA[ik], ldsB(yr, ik * 32, lane), d);
        int q = t * 128 + nt * 16 + l15;
        int rr = q >> 5, cc = q & 31;
        #pragma unroll
        for (int v = 0; v < 8; ++v) {
          int o = ot * 16 + v + 8 * hh;
          float resid = xtile[o * 16384 + rr * 128 + cc];   // exact fp32 residual
          outp[o * 1024 + q] = d[v] * scL[o] + shL[o] + resid;
        }
      }
    }
    __syncthreads();
  }
}

extern "C" void kernel_launch(void* const* d_in, const int* in_sizes, int n_in,
                              void* d_out, int out_size, void* d_ws, size_t ws_size,
                              hipStream_t stream) {
  const float* x   = (const float*)d_in[0];
  const float* tw  = (const float*)d_in[1];
  const float* tb  = (const float*)d_in[2];
  const float* pw  = (const float*)d_in[3];
  const float* pb  = (const float*)d_in[4];
  const float* gw  = (const float*)d_in[5];
  const float* gb  = (const float*)d_in[6];
  const float* ww  = (const float*)d_in[7];
  const float* wbb = (const float*)d_in[8];
  const float* bng = (const float*)d_in[9];
  const float* bnb = (const float*)d_in[10];
  const float* bnm = (const float*)d_in[11];
  const float* bnv = (const float*)d_in[12];

  (void)in_sizes; (void)n_in; (void)out_size; (void)d_ws; (void)ws_size;

  hipFuncSetAttribute(reinterpret_cast<const void*>(snlb_kernel),
                      hipFuncAttributeMaxDynamicSharedMemorySize, LDS_BYTES);

  snlb_kernel<<<dim3(128), dim3(256), LDS_BYTES, stream>>>(
      x, tw, tb, pw, pb, gw, gb, ww, wbb, bng, bnb, bnm, bnv, (float*)d_out);
}
